// GPSLayer_67276367724917
// MI455X (gfx1250) — matrix operand order repaired and checked by reading, not compiled, not run on hardware
//
#include <hip/hip_runtime.h>
#include <hip/hip_bf16.h>

typedef __attribute__((ext_vector_type(16))) _Float16 v16h;
typedef __attribute__((ext_vector_type(8)))  float    v8f;

#define N_NODES 8192
#define DIM     128
#define E_EDGES 262144
#define EDGE_D  64
#define HEADS   4
#define DH      32
#define BN_EPS  1e-5f

// ---------- ordered-float encoding for atomicMax on float via uint ----------
__device__ __forceinline__ unsigned f2o(float f) {
  unsigned u = __float_as_uint(f);
  return (u & 0x80000000u) ? ~u : (u | 0x80000000u);
}
__device__ __forceinline__ float o2f(unsigned o) {
  unsigned u = (o & 0x80000000u) ? (o & 0x7fffffffu) : ~o;
  return __uint_as_float(u);
}

// ---------- DPP16 butterfly reductions over a 16-lane half (no LDS) ----------
template <int CTRL>
__device__ __forceinline__ float dppf(float x) {
  return __int_as_float(__builtin_amdgcn_update_dpp(
      0, __float_as_int(x), CTRL, 0xF, 0xF, true));
}
__device__ __forceinline__ float rowmax16(float x) {
  x = fmaxf(x, dppf<0xB1>(x));    // quad_perm [1,0,3,2]
  x = fmaxf(x, dppf<0x4E>(x));    // quad_perm [2,3,0,1]
  x = fmaxf(x, dppf<0x141>(x));   // row_half_mirror
  x = fmaxf(x, dppf<0x140>(x));   // row_mirror
  return x;
}
__device__ __forceinline__ float rowsum16(float x) {
  x += dppf<0xB1>(x);
  x += dppf<0x4E>(x);
  x += dppf<0x141>(x);
  x += dppf<0x140>(x);
  return x;
}

// =====================================================================
// WMMA GEMM:  out[M,Nout] = A[M,K] @ W[Nout,K]^T + bias
// One wave computes a 16(M) x 64(N) strip: one A fragment feeds four
// back-to-back v_wmma_f32_16x16x32_f16 (f16 operands, f32 accumulate).
// =====================================================================
template <bool RELU, bool HALFOUT>
__global__ void gemm_bias_wmma(const float* __restrict__ A,
                               const float* __restrict__ W,
                               const float* __restrict__ bias,
                               void* __restrict__ out,
                               int M, int K, int Nout) {
  const int lane = threadIdx.x & 31;
  const int wave = threadIdx.x >> 5;
  const int mt = (blockIdx.y * (blockDim.x >> 5) + wave) * 16;
  const int nt = blockIdx.x * 64;
  if (mt >= M) return;

  const int half = lane >> 4;
  const int l16  = lane & 15;

  const float* arow  = A + (size_t)(mt + l16) * K;
  const float* wrow0 = W + (size_t)(nt + l16) * K;   // + j*16*K per sub-tile

  const int kbA = half * 16;   // A: half selects K {0..15}/{16..31}
  const int kbB = half * 16;   // B: half selects K {0..15}/{16..31}

  v8f acc[4] = {{}, {}, {}, {}};
  for (int k0 = 0; k0 < K; k0 += 32) {
    v16h a;
    const float* ap0 = arow + k0 + kbA;
    const float* ap1 = ap0 + 8;
#pragma unroll
    for (int i = 0; i < 8; ++i) {
      a[i]     = (_Float16)ap0[i];
      a[8 + i] = (_Float16)ap1[i];
    }
#pragma unroll
    for (int j = 0; j < 4; ++j) {
      v16h b;
      const float* bp = wrow0 + (size_t)j * 16 * K + k0 + kbB;
#pragma unroll
      for (int i = 0; i < 16; ++i) b[i] = (_Float16)bp[i];
      acc[j] = __builtin_amdgcn_wmma_f32_16x16x32_f16(false, a, false, b,
                                                      (short)0, acc[j],
                                                      false, false);
    }
  }

#pragma unroll
  for (int j = 0; j < 4; ++j) {
    const int col = nt + j * 16 + l16;
    const float bv = bias ? bias[col] : 0.0f;
#pragma unroll
    for (int r = 0; r < 8; ++r) {
      const int row = mt + r + half * 8;   // C layout: VGPR r -> M=r (+8 hi half)
      float vv = acc[j][r] + bv;
      if (RELU) vv = fmaxf(vv, 0.0f);
      if (HALFOUT)
        ((_Float16*)out)[(size_t)row * Nout + col] = (_Float16)vv;
      else
        ((float*)out)[(size_t)row * Nout + col] = vv;
    }
  }
}

// =====================================================================
// Edge passes (TransformerConv scatter softmax)
// =====================================================================
__global__ void edge_score_kernel(const float* __restrict__ q,
                                  const float* __restrict__ k,
                                  const _Float16* __restrict__ e16,
                                  const int* __restrict__ ei,
                                  float* __restrict__ scores,
                                  unsigned* __restrict__ m_ord) {
  const int lane = threadIdx.x & 31;
  const int e = (int)((blockIdx.x * (size_t)blockDim.x + threadIdx.x) >> 5);
  if (e >= E_EDGES) return;
  const int src = ei[e];
  const int dst = ei[E_EDGES + e];
  const float* qp = q + (size_t)dst * DIM;
  const float* kp = k + (size_t)src * DIM;
  const _Float16* ep = e16 + (size_t)e * DIM;
  float s = 0.0f;
#pragma unroll
  for (int i = 0; i < 4; ++i) {
    const int c = lane + 32 * i;
    s += qp[c] * (kp[c] + (float)ep[c]);
  }
  s = rowsum16(s);                 // DPP reduce within 16-lane halves
  s += __shfl_xor(s, 16, 32);      // combine halves
  if (lane == 0) {
    s *= 0.08838834764831845f;   // 1/sqrt(128)
    scores[e] = s;
    atomicMax(&m_ord[dst], f2o(s));
  }
}

__global__ void edge_exp_kernel(float* __restrict__ scores,   // in: score, out: exp
                                const unsigned* __restrict__ m_ord,
                                float* __restrict__ denom,
                                const int* __restrict__ ei) {
  const int e = blockIdx.x * blockDim.x + threadIdx.x;
  if (e >= E_EDGES) return;
  const int dst = ei[E_EDGES + e];
  const float ex = __expf(scores[e] - o2f(m_ord[dst]));
  atomicAdd(&denom[dst], ex);
  scores[e] = ex;
}

__global__ void edge_scatter_kernel(const float* __restrict__ v,
                                    const _Float16* __restrict__ e16,
                                    const float* __restrict__ ex,
                                    const float* __restrict__ denom,
                                    const int* __restrict__ ei,
                                    float* __restrict__ agg) {
  const int lane = threadIdx.x & 31;
  const int e = (int)((blockIdx.x * (size_t)blockDim.x + threadIdx.x) >> 5);
  if (e >= E_EDGES) return;
  const int src = ei[e];
  const int dst = ei[E_EDGES + e];
  const float alpha = ex[e] / denom[dst];
  const float* vp = v + (size_t)src * DIM;
  const _Float16* ep = e16 + (size_t)e * DIM;
  float* op = agg + (size_t)dst * DIM;
#pragma unroll
  for (int i = 0; i < 4; ++i) {
    const int c = lane + 32 * i;
    atomicAdd(&op[c], (vp[c] + (float)ep[c]) * alpha);
  }
}

// =====================================================================
// Flash-style MHA with WMMA (one wave = 16 query rows x 1 head;
// 4 waves / block).  64-key tiles: 4 score WMMAs + 4 PV WMMAs per
// iteration; row statistics via DPP16 butterflies (amortized over 64
// keys); P staged through a per-wave LDS tile for the PV A-fragments.
// qkv: [N, 3*DIM] row-major; q at col h*32, k at 128+h*32, v at 256+h*32
// =====================================================================
__global__ void flash_attn_kernel(const float* __restrict__ qkv,
                                  float* __restrict__ outO) {
  const int lane = threadIdx.x & 31;
  const int wave = threadIdx.x >> 5;
  const int half = lane >> 4;
  const int l16  = lane & 15;
  const int head = blockIdx.y;
  const int q0   = (blockIdx.x * 4 + wave) * 16;
  const int LD = 3 * DIM;
  const int qoff = head * DH;
  const int koff = DIM + head * DH;
  const int voff = 2 * DIM + head * DH;
  const float scale = 0.17677669529663687f;   // 1/sqrt(32)

  // Q tile in A-matrix layout (16 x 32), scale folded in
  v16h aQ;
  {
    const float* qp = qkv + (size_t)(q0 + l16) * LD + qoff;
    const int kb = half * 16;
#pragma unroll
    for (int i = 0; i < 8; ++i) {
      aQ[i]     = (_Float16)(qp[kb + i] * scale);
      aQ[8 + i] = (_Float16)(qp[kb + 8 + i] * scale);
    }
  }

  float mrun[8], lsum[8];
  v8f o0 = {}, o1 = {};
#pragma unroll
  for (int r = 0; r < 8; ++r) { mrun[r] = -3.0e38f; lsum[r] = 0.0f; }

  __shared__ _Float16 pShared[4][16 * 64];
  _Float16* pTile = pShared[wave];

  const int kb2 = half * 16;

  for (int j0 = 0; j0 < N_NODES; j0 += 64) {
    // 4 x (32x16) K tiles in B-matrix layout: B[d, j] = K[j0+t*16+j, d]
    v8f s[4];
    {
      v8f z = {};
#pragma unroll
      for (int t = 0; t < 4; ++t) {
        v16h bK;
        const float* kp = qkv + (size_t)(j0 + t * 16 + l16) * LD + koff + kb2;
#pragma unroll
        for (int i = 0; i < 16; ++i) bK[i] = (_Float16)kp[i];
        s[t] = __builtin_amdgcn_wmma_f32_16x16x32_f16(false, aQ, false, bK,
                                                      (short)0, z, false, false);
      }
    }

    // online softmax per row (row r+8*half lives across one 16-lane half)
#pragma unroll
    for (int r = 0; r < 8; ++r) {
      const float tm = rowmax16(fmaxf(fmaxf(s[0][r], s[1][r]),
                                      fmaxf(s[2][r], s[3][r])));
      const float nm   = fmaxf(mrun[r], tm);
      const float corr = __expf(mrun[r] - nm);
      const float p0 = __expf(s[0][r] - nm);
      const float p1 = __expf(s[1][r] - nm);
      const float p2 = __expf(s[2][r] - nm);
      const float p3 = __expf(s[3][r] - nm);
      const float rs = rowsum16((p0 + p1) + (p2 + p3));
      lsum[r] = lsum[r] * corr + rs;
      mrun[r] = nm;
      o0[r] *= corr;
      o1[r] *= corr;
      const int row = r + half * 8;
      pTile[row * 64 + l16]      = (_Float16)p0;
      pTile[row * 64 + 16 + l16] = (_Float16)p1;
      pTile[row * 64 + 32 + l16] = (_Float16)p2;
      pTile[row * 64 + 48 + l16] = (_Float16)p3;
    }
    __syncthreads();

    // reload P in A-matrix layout: aP0 = keys 0..31, aP1 = keys 32..63
    v16h aP0, aP1;
    {
      const int kb = half * 16;
      const _Float16* pp = pTile + l16 * 64;
#pragma unroll
      for (int i = 0; i < 8; ++i) {
        aP0[i]     = pp[kb + i];
        aP0[8 + i] = pp[kb + 8 + i];
        aP1[i]     = pp[32 + kb + i];
        aP1[8 + i] = pp[40 + kb + i];
      }
    }
    __syncthreads();

    // V tiles in B-matrix layout (32 keys x 16 dh), 2 key groups x 2 dh halves
#pragma unroll
    for (int g = 0; g < 2; ++g) {
      v16h bV0, bV1;
#pragma unroll
      for (int i = 0; i < 16; ++i) {
        const float* vp = qkv + (size_t)(j0 + g * 32 + kb2 + i) * LD + voff + l16;
        bV0[i] = (_Float16)vp[0];
        bV1[i] = (_Float16)vp[16];
      }
      const v16h aP = g ? aP1 : aP0;
      o0 = __builtin_amdgcn_wmma_f32_16x16x32_f16(false, aP, false, bV0,
                                                  (short)0, o0, false, false);
      o1 = __builtin_amdgcn_wmma_f32_16x16x32_f16(false, aP, false, bV1,
                                                  (short)0, o1, false, false);
    }
  }

#pragma unroll
  for (int r = 0; r < 8; ++r) {
    const int row = q0 + r + half * 8;
    const float inv = 1.0f / lsum[r];
    outO[(size_t)row * DIM + head * DH + l16]      = o0[r] * inv;
    outO[(size_t)row * DIM + head * DH + 16 + l16] = o1[r] * inv;
  }
}

// =====================================================================
// BatchNorm over node axis: stats of (a+b) per column, then apply
// =====================================================================
__global__ void addbn_stats_kernel(const float* __restrict__ a,
                                   const float* __restrict__ b,
                                   float* __restrict__ mu,
                                   float* __restrict__ var,
                                   int n, int d) {
  const int col = blockIdx.x;
  float s = 0.0f, s2 = 0.0f;
  for (int r = threadIdx.x; r < n; r += blockDim.x) {
    const float t = a[(size_t)r * d + col] + b[(size_t)r * d + col];
    s += t; s2 += t * t;
  }
  __shared__ float ls[256], ls2[256];
  ls[threadIdx.x] = s; ls2[threadIdx.x] = s2;
  __syncthreads();
  for (int off = 128; off >= 1; off >>= 1) {
    if ((int)threadIdx.x < off) {
      ls[threadIdx.x]  += ls[threadIdx.x + off];
      ls2[threadIdx.x] += ls2[threadIdx.x + off];
    }
    __syncthreads();
  }
  if (threadIdx.x == 0) {
    const float m = ls[0] / (float)n;
    mu[col] = m;
    var[col] = ls2[0] / (float)n - m * m;
  }
}

__global__ void addbn_apply_kernel(const float* __restrict__ a,
                                   const float* __restrict__ b,
                                   const float* __restrict__ g,
                                   const float* __restrict__ be,
                                   const float* __restrict__ mu,
                                   const float* __restrict__ var,
                                   float* __restrict__ out, int total, int d) {
  const int i = blockIdx.x * blockDim.x + threadIdx.x;
  if (i >= total) return;
  const int c = i & (d - 1);
  const float t = a[i] + b[i];
  out[i] = g[c] * (t - mu[c]) * rsqrtf(var[c] + BN_EPS) + be[c];
}

__global__ void add2_kernel(const float* __restrict__ a,
                            const float* __restrict__ b,
                            float* __restrict__ out, int total) {
  const int i = blockIdx.x * blockDim.x + threadIdx.x;
  if (i < total) out[i] = a[i] + b[i];
}

// =====================================================================
extern "C" void kernel_launch(void* const* d_in, const int* in_sizes, int n_in,
                              void* d_out, int out_size, void* d_ws, size_t ws_size,
                              hipStream_t stream) {
  const float* x         = (const float*)d_in[0];
  const int*   ei        = (const int*)  d_in[1];
  const float* edge_attr = (const float*)d_in[2];
  const float* Wq  = (const float*)d_in[3];  const float* bq  = (const float*)d_in[4];
  const float* Wk  = (const float*)d_in[5];  const float* bk  = (const float*)d_in[6];
  const float* Wv  = (const float*)d_in[7];  const float* bv  = (const float*)d_in[8];
  const float* We  = (const float*)d_in[9];  const float* be  = (const float*)d_in[10];
  const float* Wsk = (const float*)d_in[11]; const float* bsk = (const float*)d_in[12];
  const float* inW = (const float*)d_in[13]; const float* inB = (const float*)d_in[14];
  const float* outW= (const float*)d_in[15]; const float* outB= (const float*)d_in[16];
  const float* g1l = (const float*)d_in[17]; const float* b1l = (const float*)d_in[18];
  const float* g1a = (const float*)d_in[19]; const float* b1a = (const float*)d_in[20];
  const float* g2  = (const float*)d_in[21]; const float* b2  = (const float*)d_in[22];
  const float* W1  = (const float*)d_in[23]; const float* b1f = (const float*)d_in[24];
  const float* W2  = (const float*)d_in[25]; const float* b2f = (const float*)d_in[26];
  float* out = (float*)d_out;

  // ---- workspace carve-up (256B aligned) ----
  char* base = (char*)d_ws;
  size_t off = 0;
  auto carve = [&](size_t bytes) -> char* {
    char* p = base + off;
    off = (off + bytes + 255) & ~(size_t)255;
    return p;
  };
  const size_t ND4 = (size_t)N_NODES * DIM * 4;
  float*    q      = (float*)   carve(ND4);
  float*    k      = (float*)   carve(ND4);
  float*    v      = (float*)   carve(ND4);
  float*    hl     = (float*)   carve(ND4);                   // skip + scattered agg
  float*    qkv    = (float*)   carve((size_t)N_NODES * 3 * DIM * 4);
  float*    attnO  = (float*)   carve(ND4);
  float*    ha     = (float*)   carve(ND4);
  float*    hlbn   = (float*)   carve(ND4);
  float*    habn   = (float*)   carve(ND4);
  float*    hsum   = (float*)   carve(ND4);
  float*    ff1    = (float*)   carve((size_t)N_NODES * 2 * DIM * 4);
  float*    ff2    = (float*)   carve(ND4);
  _Float16* e16    = (_Float16*)carve((size_t)E_EDGES * DIM * 2);
  float*    scores = (float*)   carve((size_t)E_EDGES * 4);
  unsigned* m_ord  = (unsigned*)carve((size_t)N_NODES * 4);
  float*    denom  = (float*)   carve((size_t)N_NODES * 4);
  float*    mu     = (float*)   carve(DIM * 4);
  float*    var    = (float*)   carve(DIM * 4);
  (void)ws_size; (void)n_in; (void)in_sizes; (void)out_size;

  hipMemsetAsync(m_ord, 0, (size_t)N_NODES * 4, stream);
  hipMemsetAsync(denom, 0, (size_t)N_NODES * 4, stream);

  const dim3 blk256(256);
  // GEMM grids: (Nout/64, M/128), 8 waves/block, 16x64 per wave
  const dim3 gN128((DIM)/64,   N_NODES/128);
  const dim3 gQKV((3*DIM)/64,  N_NODES/128);
  const dim3 gFF1((2*DIM)/64,  N_NODES/128);
  const dim3 gE  ((DIM)/64,    E_EDGES/128);

  // --- local MPNN branch projections ---
  gemm_bias_wmma<false,false><<<gN128, blk256, 0, stream>>>(x, Wq,  bq,  q,  N_NODES, DIM, DIM);
  gemm_bias_wmma<false,false><<<gN128, blk256, 0, stream>>>(x, Wk,  bk,  k,  N_NODES, DIM, DIM);
  gemm_bias_wmma<false,false><<<gN128, blk256, 0, stream>>>(x, Wv,  bv,  v,  N_NODES, DIM, DIM);
  gemm_bias_wmma<false,false><<<gN128, blk256, 0, stream>>>(x, Wsk, bsk, hl, N_NODES, DIM, DIM);
  gemm_bias_wmma<false,true ><<<gE,   blk256, 0, stream>>>(edge_attr, We, be, e16, E_EDGES, EDGE_D, DIM);

  // --- edge scatter softmax + aggregation (into hl, on top of skip) ---
  edge_score_kernel  <<<E_EDGES/8,   blk256, 0, stream>>>(q, k, e16, ei, scores, m_ord);
  edge_exp_kernel    <<<E_EDGES/256, blk256, 0, stream>>>(scores, m_ord, denom, ei);
  edge_scatter_kernel<<<E_EDGES/8,   blk256, 0, stream>>>(v, e16, scores, denom, ei, hl);

  // --- BN(local) on x + hl ---
  addbn_stats_kernel<<<DIM, blk256, 0, stream>>>(x, hl, mu, var, N_NODES, DIM);
  addbn_apply_kernel<<<(N_NODES*DIM)/256, blk256, 0, stream>>>(x, hl, g1l, b1l, mu, var, hlbn,
                                                               N_NODES*DIM, DIM);

  // --- global attention branch ---
  gemm_bias_wmma<false,false><<<gQKV, blk256, 0, stream>>>(x, inW, inB, qkv, N_NODES, DIM, 3*DIM);
  flash_attn_kernel<<<dim3(N_NODES/64, HEADS), dim3(128), 0, stream>>>(qkv, attnO);
  gemm_bias_wmma<false,false><<<gN128, blk256, 0, stream>>>(attnO, outW, outB, ha, N_NODES, DIM, DIM);
  addbn_stats_kernel<<<DIM, blk256, 0, stream>>>(x, ha, mu, var, N_NODES, DIM);
  addbn_apply_kernel<<<(N_NODES*DIM)/256, blk256, 0, stream>>>(x, ha, g1a, b1a, mu, var, habn,
                                                               N_NODES*DIM, DIM);

  // --- combine + FFN + final BN ---
  add2_kernel<<<(N_NODES*DIM)/256, blk256, 0, stream>>>(hlbn, habn, hsum, N_NODES*DIM);
  gemm_bias_wmma<true ,false><<<gFF1, blk256, 0, stream>>>(hsum, W1, b1f, ff1, N_NODES, DIM, 2*DIM);
  gemm_bias_wmma<false,false><<<gN128, blk256, 0, stream>>>(ff1, W2, b2f, ff2, N_NODES, 2*DIM, DIM);
  addbn_stats_kernel<<<DIM, blk256, 0, stream>>>(hsum, ff2, mu, var, N_NODES, DIM);
  addbn_apply_kernel<<<(N_NODES*DIM)/256, blk256, 0, stream>>>(hsum, ff2, g2, b2, mu, var, out,
                                                               N_NODES*DIM, DIM);
}
